// MultiHeadAttention_42202348650836
// MI455X (gfx1250) — matrix operand order repaired and checked by reading, not compiled, not run on hardware
//
#include <hip/hip_runtime.h>
#include <hip/hip_bf16.h>

typedef __attribute__((ext_vector_type(16))) __bf16 v16bf;
typedef __attribute__((ext_vector_type(8)))  float  v8f;
typedef __attribute__((ext_vector_type(4)))  int    v4i;

namespace {
constexpr int NB = 4;       // batch
constexpr int NT = 2048;    // sequence length
constexpr int ND = 1024;    // model dim
constexpr int NH = 16;      // heads
constexpr int NE = 64;      // head dim
constexpr int HE = NH * NE; // 1024
// attention scale folded into the Q projection, in exp2 domain:
// (1/sqrt(64)) * log2(e)
constexpr float QSCALE = 0.18033688011112042f;
}

__device__ __forceinline__ v8f wmma_bf16(v16bf a, v16bf b, v8f c) {
  // D = A(16x32 bf16) * B(32x16 bf16) + C(16x16 f32)
  return __builtin_amdgcn_wmma_f32_16x16x32_bf16(false, a, false, b, (short)0, c,
                                                 false, false);
}

// fast exp2 -> v_exp_f32
__device__ __forceinline__ float fexp2(float x) {
#if __has_builtin(__builtin_amdgcn_exp2f)
  return __builtin_amdgcn_exp2f(x);
#else
  return exp2f(x);
#endif
}

// swap values between the two 16-lane halves (lane L <-> L^16)
__device__ __forceinline__ float permx16(float x) {
#if __has_builtin(__builtin_amdgcn_permlanex16)
  int v = __float_as_int(x);
  return __int_as_float(
      __builtin_amdgcn_permlanex16(v, v, 0x76543210, 0xfedcba98, false, false));
#else
  return __shfl_xor(x, 16, 32);
#endif
}

// ---- CDNA5 async global->LDS copy (ASYNCcnt), with safe fallback ------------
typedef __attribute__((address_space(1))) v4i* gas_v4i_p;
typedef __attribute__((address_space(3))) v4i* las_v4i_p;

__device__ __forceinline__ void async_b128(const __bf16* g, __bf16* l) {
#if __has_builtin(__builtin_amdgcn_global_load_async_to_lds_b128)
  __builtin_amdgcn_global_load_async_to_lds_b128((gas_v4i_p)g, (las_v4i_p)l, 0, 0);
#else
  uint4 t = *(const uint4*)g;
  *(uint4*)l = t;
#endif
}

template <int N>
__device__ __forceinline__ void wait_async() {
#if __has_builtin(__builtin_amdgcn_s_wait_asynccnt)
  __builtin_amdgcn_s_wait_asynccnt((unsigned short)N);
#else
  if (N == 0)
    asm volatile("s_wait_asynccnt 0x0" ::: "memory");
  else
    asm volatile("s_wait_asynccnt 0x4" ::: "memory");
#endif
}

// ---------------- weight prep: f32 -> bf16, transposed for B-fragment reads --
// W [NH][ND][NE] -> WT [NH][NE][ND]
__global__ void cvt_w_hde_to_hed(const float* __restrict__ W, __bf16* __restrict__ WT) {
  int idx = blockIdx.x * blockDim.x + threadIdx.x;  // NH*ND*NE = 1M
  int h = idx >> 16;
  int rem = idx & 65535;
  int d = rem >> 6;
  int e = rem & 63;
  WT[((size_t)h * NE + e) * ND + d] = (__bf16)W[idx];
}

// Wo [HE][ND] -> WoT [ND][HE]
__global__ void cvt_wo_T(const float* __restrict__ Wo, __bf16* __restrict__ WoT) {
  int idx = blockIdx.x * blockDim.x + threadIdx.x;  // HE*ND = 1M
  int r = idx >> 10;   // he
  int c = idx & 1023;  // d
  WoT[(size_t)c * HE + r] = (__bf16)Wo[idx];
}

// ---------------- per-head projection GEMM -----------------------------------
// OUT[b,h,t,e] = oscale * sum_d X[b,t,d] * W[h,d,e]  (W transposed [h,e,d])
template <bool TRANSPOSE_OUT>
__global__ __launch_bounds__(128) void proj_kernel(const float* __restrict__ X,
                                                   const __bf16* __restrict__ WT,
                                                   __bf16* __restrict__ OUT,
                                                   float oscale) {
  int tile = blockIdx.x;       // NB*NH*(NT/64) = 2048
  int tt = tile & 31;
  int bh = tile >> 5;
  int h = bh & (NH - 1);
  int b = bh >> 4;
  int wave = threadIdx.x >> 5;
  int lane = threadIdx.x & 31;
  int mr = lane & 15;
  int hi = (lane >> 4) & 1;
  int abase = hi ? 16 : 0;     // A layout: K = abase + i
  int bbase = hi ? 16 : 0;     // B layout: K = bbase + i
  int t0 = tt * 64 + wave * 16;

  const float* Xr = X + ((size_t)b * NT + t0 + mr) * ND;
  const __bf16* Wh = WT + (size_t)h * NE * ND;

  v8f c[4] = {};
  for (int k0 = 0; k0 < ND; k0 += 32) {
    v16bf a;
#pragma unroll
    for (int i = 0; i < 8; ++i) {
      a[i]     = (__bf16)Xr[k0 + abase + i];
      a[i + 8] = (__bf16)Xr[k0 + abase + 8 + i];
    }
#pragma unroll
    for (int n = 0; n < 4; ++n) {
      const __bf16* Wr = Wh + (size_t)(n * 16 + mr) * ND + k0 + bbase;
      v16bf bb;
#pragma unroll
      for (int i = 0; i < 16; ++i) bb[i] = Wr[i];
      c[n] = wmma_bf16(a, bb, c[n]);
    }
  }

  if (!TRANSPOSE_OUT) {
    __bf16* Or = OUT + (((size_t)b * NH + h) * NT + t0) * NE;
#pragma unroll
    for (int v = 0; v < 8; ++v) {
      int row = v + (hi ? 8 : 0);
#pragma unroll
      for (int n = 0; n < 4; ++n)
        Or[(size_t)row * NE + n * 16 + mr] = (__bf16)(c[n][v] * oscale);
    }
  } else {  // OUT[b,h,e,t] for V (so attention V-fragments are contiguous)
    __bf16* Or = OUT + ((size_t)b * NH + h) * (size_t)NE * NT;
#pragma unroll
    for (int v = 0; v < 8; ++v) {
      int row = t0 + v + (hi ? 8 : 0);
#pragma unroll
      for (int n = 0; n < 4; ++n)
        Or[(size_t)(n * 16 + mr) * NT + row] = (__bf16)(c[n][v] * oscale);
    }
  }
}

// ---------------- flash attention core ----------------------------------------
// Computes S^T = Kh*Qh^T per 32-key tile (lane owns one query column), online
// softmax in exp2 domain (scale pre-folded into Qh), O^T = Vh^T * P^T with the
// P^T B-fragment built in registers via permlanex16 (no LDS round-trip).
__global__ __launch_bounds__(128) void attn_kernel(const __bf16* __restrict__ Qh,
                                                   const __bf16* __restrict__ Kh,
                                                   const __bf16* __restrict__ VhT,
                                                   __bf16* __restrict__ Oattn) {
  constexpr int KROW = 72;  // 32x64 K tile, row padded: 144B stride
  constexpr int VROW = 40;  // 64x32 V tile (e-major), row padded: 80B stride
  __shared__ __bf16 Kt[2][32 * KROW];
  __shared__ __bf16 Vt[2][64 * VROW];

  int tile = blockIdx.x;  // NB*NH*(NT/64) = 2048
  int qt = tile & 31;
  int bh = tile >> 5;
  int h = bh & (NH - 1);
  int b = bh >> 4;
  int tid = threadIdx.x;
  int wave = tid >> 5;
  int lane = tid & 31;
  int mr = lane & 15;
  int hi = (lane >> 4) & 1;
  int abase = hi ? 16 : 0;
  int bbase = hi ? 16 : 0;
  int q0 = qt * 64 + wave * 16;

  size_t headT = ((size_t)b * NH + h) * NT;
  const __bf16* Kbase = Kh + headT * NE;
  const __bf16* Vbase = VhT + ((size_t)b * NH + h) * (size_t)NE * NT;

  // staging assignment: 128 threads x 2 chunks of 16B per tile per tensor
  int kr0 = tid >> 3, kc0 = (tid & 7) * 8;  // K: rows 0..15 (+16 for chunk 1)
  int vr0 = tid >> 2, vc0 = (tid & 3) * 8;  // V: rows 0..31 (+32 for chunk 1)

  auto stage = [&](int j0, int buf) {
    const __bf16* gk = Kbase + (size_t)(j0 + kr0) * NE + kc0;
    async_b128(gk, &Kt[buf][kr0 * KROW + kc0]);
    async_b128(gk + (size_t)16 * NE, &Kt[buf][(kr0 + 16) * KROW + kc0]);
    const __bf16* gv = Vbase + (size_t)vr0 * NT + j0 + vc0;
    async_b128(gv, &Vt[buf][vr0 * VROW + vc0]);
    async_b128(gv + (size_t)32 * NT, &Vt[buf][(vr0 + 32) * VROW + vc0]);
  };

  // Q^T B-fragments (resident): lane's column is query q0+mr
  const __bf16* Qr = Qh + (headT + q0 + mr) * NE;
  v16bf bq0, bq1;
#pragma unroll
  for (int i = 0; i < 16; ++i) {
    bq0[i] = Qr[bbase + i];
    bq1[i] = Qr[32 + bbase + i];
  }

  float mst = -3.0e38f, lst = 0.0f;  // per-lane (per-query) softmax state
  v8f oc[4] = {};                    // O^T accumulators (M=e rows, N=query)

  stage(0, 0);
  for (int jt = 0; jt < NT / 32; ++jt) {
    int j0 = jt * 32;
    int buf = jt & 1;
    if (jt + 1 < NT / 32) {
      stage(j0 + 32, buf ^ 1);  // prefetch next tile while this one computes
      wait_async<4>();          // previous tile's 4 asyncs have landed
    } else {
      wait_async<0>();
    }
    __syncthreads();

    const __bf16* Kb = &Kt[buf][0];
    const __bf16* Vb = &Vt[buf][0];

    // S^T = K * Q^T : two 16-key sub-tiles (A = K rows from LDS)
    v8f sT0 = {}, sT1 = {};
    {
      v16bf ak;
      const __bf16* kr = Kb + mr * KROW;
#pragma unroll
      for (int i = 0; i < 8; ++i) {
        ak[i] = kr[abase + i];
        ak[i + 8] = kr[abase + 8 + i];
      }
      sT0 = wmma_bf16(ak, bq0, sT0);
#pragma unroll
      for (int i = 0; i < 8; ++i) {
        ak[i] = kr[32 + abase + i];
        ak[i + 8] = kr[32 + abase + 8 + i];
      }
      sT0 = wmma_bf16(ak, bq1, sT0);
      const __bf16* kr1 = Kb + (16 + mr) * KROW;
#pragma unroll
      for (int i = 0; i < 8; ++i) {
        ak[i] = kr1[abase + i];
        ak[i + 8] = kr1[abase + 8 + i];
      }
      sT1 = wmma_bf16(ak, bq0, sT1);
#pragma unroll
      for (int i = 0; i < 8; ++i) {
        ak[i] = kr1[32 + abase + i];
        ak[i + 8] = kr1[32 + abase + 8 + i];
      }
      sT1 = wmma_bf16(ak, bq1, sT1);
    }

    // per-query online softmax, exp2 domain. Lane holds 16 of the tile's 32
    // key scores for query q0+mr; the other 16 live in the paired lane.
    float sv[16];
#pragma unroll
    for (int v = 0; v < 8; ++v) { sv[v] = sT0[v]; sv[8 + v] = sT1[v]; }

    float tm[16];
#pragma unroll
    for (int i = 0; i < 16; ++i) tm[i] = sv[i];
#pragma unroll
    for (int st = 1; st < 16; st <<= 1)
#pragma unroll
      for (int i = 0; i < 16; i += 2 * st) tm[i] = fmaxf(tm[i], tm[i + st]);
    float rm = tm[0];
    float mn = fmaxf(mst, fmaxf(rm, permx16(rm)));
    float alpha = fexp2(mst - mn);
    mst = mn;

    float p[16];
#pragma unroll
    for (int i = 0; i < 16; ++i) p[i] = fexp2(sv[i] - mn);

    float ts[16];
#pragma unroll
    for (int i = 0; i < 16; ++i) ts[i] = p[i];
#pragma unroll
    for (int st = 1; st < 16; st <<= 1)
#pragma unroll
      for (int i = 0; i < 16; i += 2 * st) ts[i] += ts[i + st];
    float rs = ts[0];
    lst = lst * alpha + rs + permx16(rs);

#pragma unroll
    for (int n = 0; n < 4; ++n)
#pragma unroll
      for (int v = 0; v < 8; ++v) oc[n][v] *= alpha;

    // build P^T B-fragment in registers:
    //   low lanes need partner's sub0 values (keys 8..15)
    //   high lanes need partner's sub1 values (keys 16..23)
    v16bf bp;
#pragma unroll
    for (int v = 0; v < 8; ++v) {
      float xs = hi ? p[v] : p[8 + v];
      float xr = permx16(xs);
      bp[v]     = (__bf16)(hi ? xr : p[v]);
      bp[8 + v] = (__bf16)(hi ? p[8 + v] : xr);
    }

    // O^T += V^T * P^T  (A = V^T rows from LDS, B = P^T shared across n)
#pragma unroll
    for (int n = 0; n < 4; ++n) {
      const __bf16* vr = Vb + (size_t)(n * 16 + mr) * VROW;
      v16bf av;
#pragma unroll
      for (int i = 0; i < 8; ++i) {
        av[i] = vr[abase + i];
        av[i + 8] = vr[abase + 8 + i];
      }
      oc[n] = wmma_bf16(av, bp, oc[n]);
    }
    __syncthreads();  // all waves done with this buffer before it is re-staged
  }

  // store O^T: lane's query row, 8 contiguous e per accumulator
  float inv = __builtin_amdgcn_rcpf(lst);
  __bf16* Or = Oattn + (headT + q0 + mr) * NE + (hi ? 8 : 0);
#pragma unroll
  for (int n = 0; n < 4; ++n)
#pragma unroll
    for (int v = 0; v < 8; ++v)
      Or[n * 16 + v] = (__bf16)(oc[n][v] * inv);
}

// ---------------- output projection -------------------------------------------
// Out[b,t,d] = sum_{h,e} Attn[b,h,t,e] * Wo[h*64+e, d]  (WoT = [d][he])
__global__ __launch_bounds__(128) void outproj_kernel(const __bf16* __restrict__ Attn,
                                                      const __bf16* __restrict__ WoT,
                                                      float* __restrict__ Out) {
  int tile = blockIdx.x;           // NB*(NT/64)*(ND/64) = 2048
  int nt = tile & 15;
  int rest = tile >> 4;
  int tt = rest & 31;
  int b = rest >> 5;
  int wave = threadIdx.x >> 5;
  int lane = threadIdx.x & 31;
  int mr = lane & 15;
  int hi = (lane >> 4) & 1;
  int abase = hi ? 16 : 0;
  int bbase = hi ? 16 : 0;
  int t0 = tt * 64 + wave * 16;
  int d0 = nt * 64;

  v8f c[4] = {};
  for (int k0 = 0; k0 < HE; k0 += 32) {
    int hh = k0 >> 6;
    int e0 = k0 & 63;
    const __bf16* Ar = Attn + (((size_t)b * NH + hh) * NT + t0 + mr) * NE + e0;
    v16bf a;
#pragma unroll
    for (int i = 0; i < 8; ++i) {
      a[i]     = Ar[abase + i];
      a[i + 8] = Ar[abase + 8 + i];
    }
#pragma unroll
    for (int n = 0; n < 4; ++n) {
      const __bf16* Wr = WoT + (size_t)(d0 + n * 16 + mr) * HE + k0 + bbase;
      v16bf bb;
#pragma unroll
      for (int i = 0; i < 16; ++i) bb[i] = Wr[i];
      c[n] = wmma_bf16(a, bb, c[n]);
    }
  }

#pragma unroll
  for (int v = 0; v < 8; ++v) {
    int row = t0 + v + (hi ? 8 : 0);
#pragma unroll
    for (int n = 0; n < 4; ++n)
      Out[((size_t)b * NT + row) * ND + d0 + n * 16 + mr] = c[n][v];
  }
}

// ---------------- host side ----------------------------------------------------
extern "C" void kernel_launch(void* const* d_in, const int* in_sizes, int n_in,
                              void* d_out, int out_size, void* d_ws, size_t ws_size,
                              hipStream_t stream) {
  (void)in_sizes; (void)n_in; (void)out_size; (void)ws_size;
  const float* Q  = (const float*)d_in[0];
  const float* K  = (const float*)d_in[1];
  const float* V  = (const float*)d_in[2];
  const float* Wq = (const float*)d_in[3];
  const float* Wk = (const float*)d_in[4];
  const float* Wv = (const float*)d_in[5];
  const float* Wo = (const float*)d_in[6];

  char* w = (char*)d_ws;
  const size_t wbytes = (size_t)NH * NE * ND * sizeof(__bf16);       // 2 MB
  const size_t abytes = (size_t)NB * NH * NT * NE * sizeof(__bf16);  // 16 MB
  __bf16* WqT = (__bf16*)w;  w += wbytes;
  __bf16* WkT = (__bf16*)w;  w += wbytes;
  __bf16* WvT = (__bf16*)w;  w += wbytes;
  __bf16* WoT = (__bf16*)w;  w += (size_t)ND * HE * sizeof(__bf16);
  __bf16* Qh  = (__bf16*)w;  w += abytes;
  __bf16* Kh  = (__bf16*)w;  w += abytes;
  __bf16* VhT = (__bf16*)w;  w += abytes;
  __bf16* At  = (__bf16*)w;  w += abytes;

  cvt_w_hde_to_hed<<<4096, 256, 0, stream>>>(Wq, WqT);
  cvt_w_hde_to_hed<<<4096, 256, 0, stream>>>(Wk, WkT);
  cvt_w_hde_to_hed<<<4096, 256, 0, stream>>>(Wv, WvT);
  cvt_wo_T<<<4096, 256, 0, stream>>>(Wo, WoT);

  // attention scale (1/8) and log2(e) folded into Qh
  proj_kernel<false><<<NB * NH * (NT / 64), 128, 0, stream>>>(Q, WqT, Qh, QSCALE);
  proj_kernel<false><<<NB * NH * (NT / 64), 128, 0, stream>>>(K, WkT, Kh, 1.0f);
  proj_kernel<true><<<NB * NH * (NT / 64), 128, 0, stream>>>(V, WvT, VhT, 1.0f);

  attn_kernel<<<NB * NH * (NT / 64), 128, 0, stream>>>(Qh, Kh, VhT, At);

  outproj_kernel<<<NB * (NT / 64) * (ND / 64), 128, 0, stream>>>(At, WoT,
                                                                 (float*)d_out);
}
